// MultiHeadAttentionTorch_9251359556149
// MI455X (gfx1250) — matrix operand order repaired and checked by reading, not compiled, not run on hardware
//
#include <hip/hip_runtime.h>

// ---------------------------------------------------------------------------
// MultiHeadAttention for MI455X (gfx1250): bf16 WMMA, flash-style streaming
// softmax with DPP16 reductions, double-buffered async global->LDS staging.
// ---------------------------------------------------------------------------

typedef __attribute__((ext_vector_type(16))) __bf16 v16bf;
typedef __attribute__((ext_vector_type(8)))  __bf16 v8bf;
typedef __attribute__((ext_vector_type(8)))  unsigned short v8us;
typedef __attribute__((ext_vector_type(8)))  float  v8f;

#define B_ 2
#define S_ 4096
#define E_ 512
#define H_ 8
#define D_ 64

// fp32 -> bf16 (RNE) as raw u16
__device__ __forceinline__ unsigned short f2bfu(float f) {
  union { float f; unsigned u; } x; x.f = f;
  unsigned r = x.u + 0x7FFFu + ((x.u >> 16) & 1u);
  return (unsigned short)(r >> 16);
}

// two fp32 -> packed bf16x2 (low = a, high = b)
__device__ __forceinline__ unsigned pk2bf(float a, float b) {
#if __has_builtin(__builtin_amdgcn_cvt_pk_bf16_f32)
  auto r = __builtin_amdgcn_cvt_pk_bf16_f32(a, b);
  return __builtin_bit_cast(unsigned, r);
#else
  return (unsigned)f2bfu(a) | ((unsigned)f2bfu(b) << 16);
#endif
}

__device__ __forceinline__ v16bf cat8(v8bf lo, v8bf hi) {
  return __builtin_shufflevector(lo, hi, 0,1,2,3,4,5,6,7,8,9,10,11,12,13,14,15);
}

__device__ __forceinline__ v8bf ld8(const unsigned short* p) {
  return *(const v8bf*)p;
}

// ---- 16-lane butterfly reductions on DPP16 (stay inside each 16-lane row,
// matching the WMMA C-layout halves; no LDS/bpermute traffic) ----
__device__ __forceinline__ float dpp_max16(float x) {
  union { float f; int i; } a, b; a.f = x;
  b.i = __builtin_amdgcn_update_dpp(0, a.i, 0xB1, 0xF, 0xF, true);  a.f = fmaxf(a.f, b.f); // quad_perm 1,0,3,2
  b.i = __builtin_amdgcn_update_dpp(0, a.i, 0x4E, 0xF, 0xF, true);  a.f = fmaxf(a.f, b.f); // quad_perm 2,3,0,1
  b.i = __builtin_amdgcn_update_dpp(0, a.i, 0x141, 0xF, 0xF, true); a.f = fmaxf(a.f, b.f); // row_half_mirror
  b.i = __builtin_amdgcn_update_dpp(0, a.i, 0x140, 0xF, 0xF, true); a.f = fmaxf(a.f, b.f); // row_mirror
  return a.f;
}
__device__ __forceinline__ float dpp_sum16(float x) {
  union { float f; int i; } a, b; a.f = x;
  b.i = __builtin_amdgcn_update_dpp(0, a.i, 0xB1, 0xF, 0xF, true);  a.f += b.f;
  b.i = __builtin_amdgcn_update_dpp(0, a.i, 0x4E, 0xF, 0xF, true);  a.f += b.f;
  b.i = __builtin_amdgcn_update_dpp(0, a.i, 0x141, 0xF, 0xF, true); a.f += b.f;
  b.i = __builtin_amdgcn_update_dpp(0, a.i, 0x140, 0xF, 0xF, true); a.f += b.f;
  return a.f;
}

// Async global->LDS copy (16B per lane), tracked by ASYNCcnt.
__device__ __forceinline__ void async_load_b128(unsigned ldsaddr,
                                                const void* gbase,
                                                unsigned voff) {
  asm volatile("global_load_async_to_lds_b128 %0, %1, %2 offset:0"
               :
               : "v"(ldsaddr), "v"(voff), "s"(gbase)
               : "memory");
}

__device__ __forceinline__ void wait_async() {
#if __has_builtin(__builtin_amdgcn_s_wait_asynccnt)
  __builtin_amdgcn_s_wait_asynccnt(0);
#else
  asm volatile("s_wait_asynccnt 0" ::: "memory");
#endif
}

// Low 32 bits of a generic pointer to LDS == LDS byte address (ISA 10.2).
__device__ __forceinline__ unsigned lds_addr32(const void* p) {
  return (unsigned)(unsigned long long)p;
}

// ---------------------------------------------------------------------------
// GEMM with bias: C[M,N] = A[M,K] @ W[N,K]^T + bias.  M=8192, N=K=512.
// Block tile 128x64, 8 waves (4x2), each wave 32x32; K-step 64 (2 WMMA steps).
// ---------------------------------------------------------------------------
template <bool A_F32, bool OUT_HEADS>
__global__ __launch_bounds__(256) void gemm_bias_kernel(
    const void* __restrict__ Ain, const float* __restrict__ W,
    const float* __restrict__ bias, void* __restrict__ Out) {
  __shared__ __align__(16) unsigned short Al[128 * 80];  // [row][k(64)+pad]
  __shared__ __align__(16) unsigned short Wl[64 * 80];   // [n-row][k(64)+pad]

  const int tid = threadIdx.x;
  const int lane = tid & 31, wave = tid >> 5;
  const int wm = wave >> 1, wn = wave & 1;
  const int mlane = lane & 15, khalf = lane >> 4;
  const int m0 = blockIdx.x * 128, n0 = blockIdx.y * 64;

  v8f acc[2][2] = {};

  for (int k0 = 0; k0 < E_; k0 += 64) {
    __syncthreads();
    // ---- stage A tile (128 x 64) as bf16 ----
    if (A_F32) {
      const float* A = (const float*)Ain;
      for (int c = tid; c < 2048; c += 256) {  // float4 chunks, 16/row
        int r = c >> 4, q = c & 15;
        float4 f = *(const float4*)(A + (size_t)(m0 + r) * E_ + k0 + q * 4);
        uint2 u; u.x = pk2bf(f.x, f.y); u.y = pk2bf(f.z, f.w);
        *(uint2*)&Al[r * 80 + q * 4] = u;
      }
    } else {
      const unsigned short* A = (const unsigned short*)Ain;
      for (int c = tid; c < 1024; c += 256) {  // 16B chunks, 8/row
        int r = c >> 3, q = c & 7;
        *(v8bf*)&Al[r * 80 + q * 8] =
            *(const v8bf*)(A + (size_t)(m0 + r) * E_ + k0 + q * 8);
      }
    }
    // ---- stage W tile (64 x 64) as bf16 ----
    for (int c = tid; c < 1024; c += 256) {
      int r = c >> 4, q = c & 15;
      float4 f = *(const float4*)(W + (size_t)(n0 + r) * E_ + k0 + q * 4);
      uint2 u; u.x = pk2bf(f.x, f.y); u.y = pk2bf(f.z, f.w);
      *(uint2*)&Wl[r * 80 + q * 4] = u;
    }
    __syncthreads();

    v16bf af[2][2], bw[2][2];
#pragma unroll
    for (int mi = 0; mi < 2; ++mi)
#pragma unroll
      for (int t = 0; t < 2; ++t) {
        const unsigned short* p =
            &Al[(wm * 32 + mi * 16 + mlane) * 80 + t * 32 + khalf * 16];
        af[mi][t] = cat8(ld8(p), ld8(p + 8));
      }
#pragma unroll
    for (int ni = 0; ni < 2; ++ni)
#pragma unroll
      for (int t = 0; t < 2; ++t) {
        const unsigned short* p =
            &Wl[(wn * 32 + ni * 16 + mlane) * 80 + t * 32 + khalf * 16];
        bw[ni][t] = cat8(ld8(p), ld8(p + 8));
      }
#pragma unroll
    for (int t = 0; t < 2; ++t)
#pragma unroll
      for (int mi = 0; mi < 2; ++mi)
#pragma unroll
        for (int ni = 0; ni < 2; ++ni)
          acc[mi][ni] = __builtin_amdgcn_wmma_f32_16x16x32_bf16(
              false, af[mi][t], false, bw[ni][t], (short)0, acc[mi][ni],
              false, false);
  }

  // ---- epilogue: C/D layout row = i + 8*khalf, col = mlane ----
#pragma unroll
  for (int mi = 0; mi < 2; ++mi) {
#pragma unroll
    for (int ni = 0; ni < 2; ++ni) {
#pragma unroll
      for (int i = 0; i < 8; ++i) {
        int gm = m0 + wm * 32 + mi * 16 + i + 8 * khalf;
        int gn = n0 + wn * 32 + ni * 16 + mlane;
        float val = acc[mi][ni][i] + bias[gn];
        if (OUT_HEADS) {
          int bb = gm >> 12, s = gm & (S_ - 1);
          int hh = gn >> 6, dd = gn & (D_ - 1);
          ((unsigned short*)Out)[(((size_t)bb * H_ + hh) * S_ + s) * D_ + dd] =
              f2bfu(val);
        } else {
          ((float*)Out)[(size_t)gm * E_ + gn] = val;
        }
      }
    }
  }
}

// ---------------------------------------------------------------------------
// Flash attention core. Grid: (S/128, H, B), 256 threads (8 waves).
// Each wave owns 16 q-rows; double-buffered 64-key K/V tiles: async K DMA and
// V register-prefetch for tile j+1 overlap the WMMA/softmax of tile j.
// ---------------------------------------------------------------------------
__global__ __launch_bounds__(256) void attn_kernel(
    const unsigned short* __restrict__ Qh, const unsigned short* __restrict__ Kh,
    const unsigned short* __restrict__ Vh, unsigned short* __restrict__ Oc) {
  __shared__ __align__(16) unsigned short Kt[2][64 * 80];   // [key][d(64)+pad]
  __shared__ __align__(16) unsigned short Vt[2][64 * 80];   // [d][key(64)+pad]
  __shared__ __align__(16) unsigned short Pb[8 * 16 * 80];  // per-wave P patch

  const int tid = threadIdx.x;
  const int lane = tid & 31, wave = tid >> 5;
  const int mlane = lane & 15, khalf = lane >> 4;
  const int b = blockIdx.z, h = blockIdx.y;
  const int q0 = blockIdx.x * 128;
  const size_t bh = (size_t)b * H_ + h;
  const int qw = q0 + wave * 16;
  const float scale = 0.044194173824159216f;  // 1/sqrt(512)
  const int NJ = S_ / 64;

  // per-thread staging coordinates (2 chunks each for K and V)
  const int kr0 = tid >> 3, kq0 = tid & 7;           // chunk c = tid
  const int kr1 = (tid + 256) >> 3, kq1 = tid & 7;   // chunk c = tid + 256
  const int vk0 = kr0, vd0 = kq0 * 8;
  const int vk1 = kr1, vd1 = kq1 * 8;

  // Q A-fragments resident for the whole KV sweep (2 K-steps over D=64)
  v16bf aq[2];
  {
    const unsigned short* qrow = Qh + (bh * S_ + qw + mlane) * D_;
#pragma unroll
    for (int t = 0; t < 2; ++t) {
      const unsigned short* p = qrow + t * 32 + khalf * 16;
      aq[t] = cat8(*(const v8bf*)p, *(const v8bf*)(p + 8));
    }
  }

  v8f o[4] = {};
  float mrow[8], lrow[8];
#pragma unroll
  for (int i = 0; i < 8; ++i) { mrow[i] = -3.0e38f; lrow[i] = 0.f; }

  unsigned short* Pw = &Pb[wave * 16 * 80];

  // ---- prologue: stage tile 0 into buffer 0 ----
  {
    const unsigned short* kbase = Kh + bh * S_ * D_;  // tile 0
    const unsigned ktLds = lds_addr32(&Kt[0][0]);
    async_load_b128(ktLds + (unsigned)(kr0 * 160 + kq0 * 16), kbase,
                    (unsigned)((kr0 * D_ + kq0 * 8) * 2));
    async_load_b128(ktLds + (unsigned)(kr1 * 160 + kq1 * 16), kbase,
                    (unsigned)((kr1 * D_ + kq1 * 8) * 2));
    const unsigned short* vbase = Vh + bh * S_ * D_;
    v8us va = *(const v8us*)(vbase + vk0 * D_ + vd0);
    v8us vb = *(const v8us*)(vbase + vk1 * D_ + vd1);
#pragma unroll
    for (int i = 0; i < 8; ++i) Vt[0][(vd0 + i) * 80 + vk0] = va[i];
#pragma unroll
    for (int i = 0; i < 8; ++i) Vt[0][(vd1 + i) * 80 + vk1] = vb[i];
    wait_async();
    __syncthreads();
  }

  for (int j = 0; j < NJ; ++j) {
    const int buf = j & 1;
    const bool hasNext = (j + 1) < NJ;

    // ---- prefetch tile j+1: async K -> LDS(buf^1), V -> registers ----
    v8us va, vb;
    if (hasNext) {
      const size_t kvn = bh * S_ + (size_t)(j + 1) * 64;
      const unsigned short* kbase = Kh + kvn * D_;
      const unsigned ktLds = lds_addr32(&Kt[buf ^ 1][0]);
      async_load_b128(ktLds + (unsigned)(kr0 * 160 + kq0 * 16), kbase,
                      (unsigned)((kr0 * D_ + kq0 * 8) * 2));
      async_load_b128(ktLds + (unsigned)(kr1 * 160 + kq1 * 16), kbase,
                      (unsigned)((kr1 * D_ + kq1 * 8) * 2));
      const unsigned short* vbase = Vh + kvn * D_;
      va = *(const v8us*)(vbase + vk0 * D_ + vd0);
      vb = *(const v8us*)(vbase + vk1 * D_ + vd1);
    }

    const unsigned short* KtB = Kt[buf];
    const unsigned short* VtB = Vt[buf];

    // ---- S = Q @ K^T (2 K-steps x 4 key-groups) ----
    v8f s[4] = {};
#pragma unroll
    for (int t = 0; t < 2; ++t)
#pragma unroll
      for (int kg = 0; kg < 4; ++kg) {
        const unsigned short* p =
            &KtB[(kg * 16 + mlane) * 80 + t * 32 + khalf * 16];
        v16bf bk = cat8(ld8(p), ld8(p + 8));
        s[kg] = __builtin_amdgcn_wmma_f32_16x16x32_bf16(
            false, aq[t], false, bk, (short)0, s[kg], false, false);
      }

    // ---- online softmax (DPP16 butterflies, no LDS traffic) ----
#pragma unroll
    for (int i = 0; i < 8; ++i) {
      float v0 = s[0][i] * scale, v1 = s[1][i] * scale;
      float v2 = s[2][i] * scale, v3 = s[3][i] * scale;
      float rm = dpp_max16(fmaxf(fmaxf(v0, v1), fmaxf(v2, v3)));
      float mnew = fmaxf(mrow[i], rm);
      float corr = __expf(mrow[i] - mnew);
      float p0 = __expf(v0 - mnew), p1 = __expf(v1 - mnew);
      float p2 = __expf(v2 - mnew), p3 = __expf(v3 - mnew);
      float rs = dpp_sum16((p0 + p1) + (p2 + p3));
      lrow[i] = lrow[i] * corr + rs;
      mrow[i] = mnew;
#pragma unroll
      for (int t = 0; t < 4; ++t) o[t][i] *= corr;
      int prow = i + 8 * khalf;  // C layout -> P patch (A layout read below)
      Pw[prow * 80 + mlane] = f2bfu(p0);
      Pw[prow * 80 + 16 + mlane] = f2bfu(p1);
      Pw[prow * 80 + 32 + mlane] = f2bfu(p2);
      Pw[prow * 80 + 48 + mlane] = f2bfu(p3);
    }

    // ---- O += P @ V (K=64: 2 K-steps x 4 d-groups) ----
    v16bf ap[2];
#pragma unroll
    for (int t = 0; t < 2; ++t) {
      const unsigned short* p = &Pw[mlane * 80 + t * 32 + khalf * 16];
      ap[t] = cat8(ld8(p), ld8(p + 8));
    }
#pragma unroll
    for (int t = 0; t < 2; ++t)
#pragma unroll
      for (int dg = 0; dg < 4; ++dg) {
        const unsigned short* p =
            &VtB[(dg * 16 + mlane) * 80 + t * 32 + khalf * 16];
        v16bf bv = cat8(ld8(p), ld8(p + 8));
        o[dg] = __builtin_amdgcn_wmma_f32_16x16x32_bf16(
            false, ap[t], false, bv, (short)0, o[dg], false, false);
      }

    // ---- commit prefetched V registers to LDS(buf^1), then sync ----
    if (hasNext) {
      unsigned short* Vn = Vt[buf ^ 1];
#pragma unroll
      for (int i = 0; i < 8; ++i) Vn[(vd0 + i) * 80 + vk0] = va[i];
#pragma unroll
      for (int i = 0; i < 8; ++i) Vn[(vd1 + i) * 80 + vk1] = vb[i];
      wait_async();
    }
    __syncthreads();
  }

  // ---- normalize + write concat buffer [B,S,E] bf16 ----
#pragma unroll
  for (int i = 0; i < 8; ++i) lrow[i] = 1.0f / lrow[i];
#pragma unroll
  for (int t = 0; t < 4; ++t) {
#pragma unroll
    for (int i = 0; i < 8; ++i) {
      int row = qw + i + 8 * khalf;
      int col = h * 64 + t * 16 + mlane;
      Oc[((size_t)b * S_ + row) * E_ + col] = f2bfu(o[t][i] * lrow[i]);
    }
  }
}

// ---------------------------------------------------------------------------
extern "C" void kernel_launch(void* const* d_in, const int* in_sizes, int n_in,
                              void* d_out, int out_size, void* d_ws,
                              size_t ws_size, hipStream_t stream) {
  (void)in_sizes; (void)n_in; (void)out_size; (void)ws_size;
  const float* q  = (const float*)d_in[0];
  const float* k  = (const float*)d_in[1];
  const float* v  = (const float*)d_in[2];
  const float* Wq = (const float*)d_in[3];
  const float* bq = (const float*)d_in[4];
  const float* Wk = (const float*)d_in[5];
  const float* bk = (const float*)d_in[6];
  const float* Wv = (const float*)d_in[7];
  const float* bv = (const float*)d_in[8];
  const float* Wo = (const float*)d_in[9];
  const float* bo = (const float*)d_in[10];

  const size_t nPerHeadBuf = (size_t)B_ * H_ * S_ * D_;  // 4M bf16 elems
  unsigned short* Qh = (unsigned short*)d_ws;
  unsigned short* Kh = Qh + nPerHeadBuf;
  unsigned short* Vh = Kh + nPerHeadBuf;
  unsigned short* Oc = Vh + nPerHeadBuf;  // [B,S,E] bf16 concat

  dim3 gGemm((B_ * S_) / 128, E_ / 64);  // 64 x 8
  dim3 gAttn(S_ / 128, H_, B_);          // 32 x 8 x 2

  gemm_bias_kernel<true, true><<<gGemm, 256, 0, stream>>>(q, Wq, bq, Qh);
  gemm_bias_kernel<true, true><<<gGemm, 256, 0, stream>>>(k, Wk, bk, Kh);
  gemm_bias_kernel<true, true><<<gGemm, 256, 0, stream>>>(v, Wv, bv, Vh);
  attn_kernel<<<gAttn, 256, 0, stream>>>(Qh, Kh, Vh, Oc);
  gemm_bias_kernel<false, false><<<gGemm, 256, 0, stream>>>(Oc, Wo, bo, (float*)d_out);
}
